// DySample_76716705841930
// MI455X (gfx1250) — matrix operand order repaired and checked by reading, not compiled, not run on hardware
//
#include <hip/hip_runtime.h>
#include <hip/hip_bf16.h>
#include <cmath>

typedef __attribute__((ext_vector_type(16))) _Float16 v16h;
typedef __attribute__((ext_vector_type(8)))  float    v8f;

#define Bn 8
#define Cn 256
#define Hn 96
#define Wn 96
#define HW (Hn * Wn)          // 9216
#define OC 32                 // offset channels
#define HS (Hn * 2)           // 192
#define WS (Wn * 2)           // 192
#define TILES_PER_B (HW / 16) // 576

// ---------------------------------------------------------------------------
// Kernel A: per-pixel 1x1 conv  off[p, o] = sum_c x[b,c,p] * w[o,c]
// via v_wmma_f32_16x16x32_f16.  One wave handles a 16-pixel x 32-channel tile.
// Result stored pixel-major: offbuf[(b*HW + p) * 32 + o]  (f32).
// ---------------------------------------------------------------------------
__global__ __launch_bounds__(256) void dysample_offset_gemm(
    const float* __restrict__ x, const float* __restrict__ w,
    const float* __restrict__ bias, float* __restrict__ offbuf)
{
  const int wave = threadIdx.x >> 5;
  const int lane = threadIdx.x & 31;
  const int tile = blockIdx.x * 8 + wave;
  const int b    = tile / TILES_PER_B;
  const int p0   = (tile % TILES_PER_B) * 16;

  const int m    = lane & 15;  // A row / C,D column index
  const int half = lane >> 4;  // which K half-block this lane carries

  const float* xA = x + (size_t)b * Cn * HW + p0 + m;

  v8f acc0 = {};
  v8f acc1 = {};

  for (int k0 = 0; k0 < Cn; k0 += 32) {
    // ---- A fragment: contiguous k order (k = 16*half + e, same as B) ----
    // lane(half=0): VGPR0..7 = K 0..15
    // lane(half=1): VGPR0..7 = K 16..31
    v16h a;
#pragma unroll
    for (int e = 0; e < 16; ++e) {
      const int k = half * 16 + e;
      a[e] = (_Float16)xA[(size_t)(k0 + k) * HW];
    }
    // ---- B fragments: contiguous k order, same as A (lane = column n) ----
    // lanes 0-15 carry K=0..15, lanes 16-31 carry K=16..31 (packed pairs)
    const float* wB0 = w + (size_t)m * Cn        + k0 + half * 16;
    const float* wB1 = w + (size_t)(m + 16) * Cn + k0 + half * 16;
    v16h b0, b1;
#pragma unroll
    for (int e = 0; e < 16; ++e) {
      b0[e] = (_Float16)wB0[e];
      b1[e] = (_Float16)wB1[e];
    }
    acc0 = __builtin_amdgcn_wmma_f32_16x16x32_f16(false, a, false, b0,
                                                  (short)0, acc0, false, false);
    acc1 = __builtin_amdgcn_wmma_f32_16x16x32_f16(false, a, false, b1,
                                                  (short)0, acc1, false, false);
  }

  // ---- D layout: VGPR v -> row (v + 8*half), col = lane&15 (per 16-wide tile)
  float* outp = offbuf + ((size_t)b * HW + p0) * OC;
  const float bia0 = bias[m];
  const float bia1 = bias[m + 16];
#pragma unroll
  for (int v = 0; v < 8; ++v) {
    const int row = v + 8 * half;
    outp[(size_t)row * OC + m]      = acc0[v] + bia0;
    outp[(size_t)row * OC + m + 16] = acc1[v] + bia1;
  }
}

// ---------------------------------------------------------------------------
// Kernel B: pixel-shuffle + tanh + bilinear border grid-sample.
// One thread per (b, group, hs, ws); loops over the group's 64 channels so the
// offset/tanh/index math is done once and the 4 gather bases are reused.
// ---------------------------------------------------------------------------
__global__ __launch_bounds__(256) void dysample_sample(
    const float* __restrict__ x, const float* __restrict__ offbuf,
    float* __restrict__ out)
{
  const int PT = (HS * WS) / 256;        // 144 pixel tiles per (b, gi)
  int bid   = blockIdx.x;
  int ptile = bid % PT;
  int t     = bid / PT;
  int gi    = t & 3;
  int b     = t >> 2;

  int p  = ptile * 256 + threadIdx.x;
  int hs = p / WS;
  int ws = p - hs * WS;

  // pixel_shuffle inverse: source conv pixel + sub-channel
  int h   = hs >> 1;
  int wq  = ws >> 1;
  int sub = ((hs & 1) << 1) | (ws & 1);

  const float* op = offbuf + ((size_t)b * HW + h * Wn + wq) * OC + gi * 8 + sub;
  float offx = op[0];   // x-component channel: (2*gi)*4   + sub
  float offy = op[4];   // y-component channel: (2*gi+1)*4 + sub

  // grid -> source pixel coords (align_corners=False, +0.125*tanh, border clamp)
  float gx = 0.5f * (float)ws - 0.25f + 6.0f * tanhf(offx);
  float gy = 0.5f * (float)hs - 0.25f + 6.0f * tanhf(offy);
  gx = fminf(fmaxf(gx, 0.0f), (float)(Wn - 1));
  gy = fminf(fmaxf(gy, 0.0f), (float)(Hn - 1));

  float x0f = floorf(gx), y0f = floorf(gy);
  float fx = gx - x0f,    fy = gy - y0f;
  int x0 = (int)x0f, y0 = (int)y0f;
  int x1 = min(x0 + 1, Wn - 1);
  int y1 = min(y0 + 1, Hn - 1);

  int i00 = y0 * Wn + x0, i01 = y0 * Wn + x1;
  int i10 = y1 * Wn + x0, i11 = y1 * Wn + x1;
  float w00 = (1.0f - fx) * (1.0f - fy);
  float w01 = fx * (1.0f - fy);
  float w10 = (1.0f - fx) * fy;
  float w11 = fx * fy;

  const float* xb = x   + ((size_t)b * Cn + gi * 64) * HW;
  float*       ob = out + (((size_t)b * Cn + gi * 64) * HS + hs) * (size_t)WS + ws;

#pragma unroll 4
  for (int cc = 0; cc < 64; ++cc) {
    const float* xc = xb + (size_t)cc * HW;
    float v = xc[i00] * w00 + xc[i01] * w01 + xc[i10] * w10 + xc[i11] * w11;
    ob[(size_t)cc * HS * WS] = v;
  }
}

// ---------------------------------------------------------------------------
extern "C" void kernel_launch(void* const* d_in, const int* in_sizes, int n_in,
                              void* d_out, int out_size, void* d_ws, size_t ws_size,
                              hipStream_t stream) {
  const float* x    = (const float*)d_in[0];  // [8,256,96,96]
  const float* w    = (const float*)d_in[1];  // [32,256]
  const float* bias = (const float*)d_in[2];  // [32]
  float*       out  = (float*)d_out;          // [8,256,192,192]
  float*       offb = (float*)d_ws;           // [8*9216, 32] = 9.4 MB

  // Kernel A: 8 waves/block, one 16x32 tile per wave -> 4608 tiles / 8 = 576 blocks
  dysample_offset_gemm<<<(Bn * TILES_PER_B) / 8, 256, 0, stream>>>(x, w, bias, offb);

  // Kernel B: 8 batches * 4 groups * 144 pixel-tiles = 4608 blocks
  dysample_sample<<<Bn * 4 * ((HS * WS) / 256), 256, 0, stream>>>(x, offb, out);
}
